// LSTM_87617332839017
// MI455X (gfx1250) — compile-verified, matrix operand order checked by reading, not run on hardware
//
#include <hip/hip_runtime.h>

// LSTM: L=2, T=512, B=64, H=1024. Persistent per-layer kernel, bf16 WMMA,
// TDM (tensor_load_to_lds) staged A panels, L2-resident weights.
#define Lk 2
#define Tk 512
#define Bk 64
#define Hk 1024
#define G4H 4096            // 4*H
#define K2 2048             // H (x) + H (h) fused K dimension
#define NWG 32              // one WG per 32 hidden columns
#define WGCOLS 32
#define THREADS 256         // 8 waves (wave32)
#define HP 1032             // padded row stride (elements) in global staging AND LDS
#define PSTR HP             // LDS panel row stride == global staging row stride
#define BPH (Bk * HP)       // padded panel elements
#define BHO (Bk * Hk)       // logical (unpadded) panel elements

#define USE_TDM 1           // 1: tensor_load_to_lds (TENSORcnt); 0: per-lane async (ASYNCcnt)

typedef __attribute__((ext_vector_type(16))) __bf16 bf16x16;
typedef __attribute__((ext_vector_type(8)))  __bf16 bf16x8;
typedef __attribute__((ext_vector_type(8)))  float  v8f;
typedef __attribute__((ext_vector_type(4)))  unsigned su4;
typedef __attribute__((ext_vector_type(8)))  unsigned su8;

union BF16x16U { bf16x16 v; bf16x8 h[2]; };

#define WMMA_BF16(A, B, C) \
    __builtin_amdgcn_wmma_f32_16x16x32_bf16(false, (A), false, (B), (short)0, (C), false, false)

__device__ __forceinline__ void tensor_wait0() {
#if __has_builtin(__builtin_amdgcn_s_wait_tensorcnt)
    __builtin_amdgcn_s_wait_tensorcnt(0);
#else
    asm volatile("s_wait_tensorcnt 0x0" ::: "memory");
#endif
}
__device__ __forceinline__ void async_wait0() {
#if __has_builtin(__builtin_amdgcn_s_wait_asynccnt)
    __builtin_amdgcn_s_wait_asynccnt(0);
#else
    asm volatile("s_wait_asynccnt 0x0" ::: "memory");
#endif
}

// One TDM descriptor moves a whole [64][1032]-bf16 panel (2D tile of
// 516x64 dwords, stride 516 dwords) global -> LDS. ISA 8.3/8.4 bitfields.
__device__ __forceinline__ void tdm_load_panel(__bf16* lds_dst, const __bf16* gsrc) {
    unsigned long long ga = (unsigned long long)(uintptr_t)gsrc;
    su4 g0;
    g0[0] = 1u;                                     // count=1 (valid), user mode
    g0[1] = (unsigned)(uintptr_t)lds_dst;           // lds_addr (bytes)
    g0[2] = (unsigned)(ga & 0xFFFFFFFFu);           // global_addr[31:0]
    g0[3] = (unsigned)((ga >> 32) & 0x1FFFFFFu)     // global_addr[56:32]
          | 0x80000000u;                            // type=2 ("image")
    su8 g1;
    g1[0] = (2u << 16);                             // wg_mask=0, data_size=4B, no pad/iter
    g1[1] = (516u << 16);                           // tensor_dim0[15:0]=516 (in [31:16])
    g1[2] = (64u << 16);                            // tensor_dim0[31:16]=0 | tensor_dim1=64
    g1[3] = (516u << 16);                           // tensor_dim1 hi=0 | tile_dim0=516
    g1[4] = 64u;                                    // tile_dim1=64 | tile_dim2=0
    g1[5] = 516u;                                   // tensor_dim0_stride[31:0]=516
    g1[6] = 0u;                                     // stride0 hi | tensor_dim1_stride lo
    g1[7] = 0u;                                     // tensor_dim1_stride hi
    asm volatile("tensor_load_to_lds %0, %1" :: "s"(g0), "s"(g1) : "memory");
}

// 16 bf16 per lane = two contiguous 16B chunks (ISA 16-bit A/B layout:
// lanes 0-15 hold K in [0,8)+[16,24), lanes 16-31 hold K in [8,16)+[24,32);
// caller pre-offsets the pointer by row*ld + ((lane>>4)<<3)).
__device__ __forceinline__ bf16x16 load_b16(const __bf16* p) {
    BF16x16U u;
    u.h[0] = *(const bf16x8*)(p);
    u.h[1] = *(const bf16x8*)(p + 16);
    return u.v;
}

__device__ __forceinline__ float sigm(float x) { return 1.0f / (1.0f + __expf(-x)); }

#if !USE_TDM
// Fallback: per-lane async copy; global staging and LDS share the padded
// layout, so the panel copy is a flat BPH-element copy.
__device__ __forceinline__ void stage_panel_async(__bf16* dst, const __bf16* src, int tid) {
    for (int idx = tid; idx < BPH / 8; idx += THREADS) {
        unsigned           lds = (unsigned)(uintptr_t)(dst + idx * 8);
        unsigned long long ga  = (unsigned long long)(uintptr_t)(src + idx * 8);
        asm volatile("global_load_async_to_lds_b128 %0, %1, off"
                     :: "v"(lds), "v"(ga) : "memory");
    }
}
#endif

// K=1024 GEMM block: A-tiles from LDS panel, B-tile from global weights.
// Both register double-buffered so next-iteration loads issue before the
// current WMMAs (partial s_wait_dscnt/loadcnt instead of full drains).
__device__ __forceinline__ void gemm_lds_k1024(const __bf16* ap,   // LDS: + nl*PSTR + kch
                                               const __bf16* wp,   // global weight row
                                               v8f& c0, v8f& c1, v8f& c2, v8f& c3) {
    bf16x16 b  = load_b16(wp);
    bf16x16 a0 = load_b16(ap);
    bf16x16 a1 = load_b16(ap + 16 * PSTR);
    bf16x16 a2 = load_b16(ap + 32 * PSTR);
    bf16x16 a3 = load_b16(ap + 48 * PSTR);
#pragma unroll 2
    for (int kk = 32; kk < Hk; kk += 32) {
        bf16x16 nb = load_b16(wp + kk);
        bf16x16 n0 = load_b16(ap + kk);
        bf16x16 n1 = load_b16(ap + 16 * PSTR + kk);
        bf16x16 n2 = load_b16(ap + 32 * PSTR + kk);
        bf16x16 n3 = load_b16(ap + 48 * PSTR + kk);
        c0 = WMMA_BF16(a0, b, c0);
        c1 = WMMA_BF16(a1, b, c1);
        c2 = WMMA_BF16(a2, b, c2);
        c3 = WMMA_BF16(a3, b, c3);
        b = nb; a0 = n0; a1 = n1; a2 = n2; a3 = n3;
    }
    c0 = WMMA_BF16(a0, b, c0);
    c1 = WMMA_BF16(a1, b, c1);
    c2 = WMMA_BF16(a2, b, c2);
    c3 = WMMA_BF16(a3, b, c3);
}

__device__ __forceinline__ void grid_barrier(unsigned* counter, unsigned target) {
    __threadfence();
    __syncthreads();
    if (threadIdx.x == 0) {
        __hip_atomic_fetch_add(counter, 1u, __ATOMIC_RELEASE, __HIP_MEMORY_SCOPE_AGENT);
        while (__hip_atomic_load(counter, __ATOMIC_ACQUIRE, __HIP_MEMORY_SCOPE_AGENT) < target)
            __builtin_amdgcn_s_sleep(2);
    }
    __syncthreads();
}

// Prep: pack [Wi ; Wh] -> bf16 row-major [L][4H][2H], bsum = bi + bh, zero barriers.
__global__ void lstm_prep_kernel(const float* __restrict__ i2h_w,
                                 const float* __restrict__ i2h_b,
                                 const float* __restrict__ h2h_w,
                                 const float* __restrict__ h2h_b,
                                 __bf16* __restrict__ wcat,
                                 float* __restrict__ bsum,
                                 unsigned* __restrict__ counters) {
    size_t idx = (size_t)blockIdx.x * blockDim.x + threadIdx.x;
    const size_t total = (size_t)Lk * G4H * K2;
    if (idx < total) {
        size_t k = idx % K2;
        size_t n = (idx / K2) % G4H;
        size_t l = idx / ((size_t)K2 * G4H);
        float v = (k < Hk) ? i2h_w[(l * G4H + n) * Hk + k]
                           : h2h_w[(l * G4H + n) * Hk + (k - Hk)];
        wcat[idx] = (__bf16)v;
    }
    if (idx < (size_t)Lk * G4H) bsum[idx] = i2h_b[idx] + h2h_b[idx];
    if (idx < 2) counters[idx] = 0u;
}

// Persistent per-layer LSTM. WG owns hidden cols [32*wg, 32*wg+32).
// Wave w: gate g=w&3, column tile nt=w>>2; computes all 4 batch tiles from
// the LDS A panel, reusing the global B (weight) tile across them.
// XSTAGE: layer-0 converts f32 x -> bf16 xstage one step ahead.
// HMOD / XMASK (powers of two): time-slot wrap for h feedback / x source.
template<bool XSTAGE, bool FINAL, int HMOD, int XMASK>
__global__ __launch_bounds__(THREADS) void lstm_layer_kernel(
        const float* __restrict__ xraw,    // f32 x (layer 0 only)
        const __bf16* __restrict__ xseq,   // bf16 x source (xstage or hs0), HP-padded rows
        __bf16* __restrict__ xstage_w,     // writable xstage (layer 0 only)
        const __bf16* __restrict__ wcat,   // [4H][2H] bf16 row-major (this layer)
        const float* __restrict__ bsum,    // [4H] (this layer)
        __bf16* __restrict__ hbuf,         // h feedback buffer, HMOD slots, HP-padded rows
        float* __restrict__ dout,
        int layer,
        unsigned* __restrict__ counter) {
    const int tid  = threadIdx.x;
    const int wv   = tid >> 5;
    const int lane = tid & 31;
    const int c0   = blockIdx.x * WGCOLS;    // hidden column base
    const int g    = wv & 3;                 // gate
    const int nt   = wv >> 2;                // column tile within WG (0..1)
    const int nl   = lane & 15;
    const int hi   = lane >> 4;              // half-wave select
    const int kch  = hi << 3;                // K-chunk offset per ISA layout

    extern __shared__ __align__(16) char smem_dyn[];
    __bf16* xpan = (__bf16*)smem_dyn;                 // [64][PSTR]
    __bf16* hpan = xpan + (size_t)Bk * PSTR;          // [64][PSTR]

    __shared__ float preact[4][64][WGCOLS + 1];       // padded: no bank conflicts
    __shared__ float cst[64][WGCOLS + 1];             // per-WG private cell state
    __shared__ float bsh[4][WGCOLS];

    if (tid < 4 * WGCOLS) bsh[tid >> 5][tid & 31] = bsum[(tid >> 5) * Hk + c0 + (tid & 31)];
    for (int e = tid; e < 64 * WGCOLS; e += THREADS) cst[e >> 5][e & 31] = 0.0f;

    unsigned bar = 0;
    if (XSTAGE) {   // convert x_0 slice -> xstage slot 0 (padded rows), publish grid-wide
        for (int i = tid; i < 2048; i += THREADS) {
            int fl = blockIdx.x * 2048 + i;                  // flat in [64][1024]
            xstage_w[(fl >> 10) * HP + (fl & 1023)] = (__bf16)xraw[fl];
        }
        grid_barrier(counter, (++bar) * gridDim.x);
    } else {
        __syncthreads();
    }

    // B-tile weight row for this wave's gate / column-tile / lane column.
    const __bf16* wp = wcat + (size_t)(g * Hk + c0 + nt * 16 + nl) * K2 + kch;
    const int aoff = nl * PSTR + kch;                 // lane offset into LDS panels

    for (int t = 0; t < Tk; ++t) {
        // ---- stage A panels (global bf16 -> LDS) ----
        const __bf16* xsrc_t = xseq + (size_t)(t & XMASK) * BPH;
        const __bf16* hsrc_t = hbuf + (size_t)((t - 1) & (HMOD - 1)) * BPH;
#if USE_TDM
        if (wv == 0) {
            tdm_load_panel(xpan, xsrc_t);
            if (t > 0) tdm_load_panel(hpan, hsrc_t);
        }
#else
        stage_panel_async(xpan, xsrc_t, tid);
        if (t > 0) stage_panel_async(hpan, hsrc_t, tid);
#endif
        // prefetch this step's weight stream (branch-free, outside K loops)
        __builtin_prefetch(wp, 0, 3);
        __builtin_prefetch(wp + Hk, 0, 3);
#if USE_TDM
        if (wv == 0) tensor_wait0();
#else
        async_wait0();
#endif
        __syncthreads();

        // ---- GEMM: preact tiles = [x_t ; h_{t-1}] @ Wcat^T ----
        v8f acc0 = {}, acc1 = {}, acc2 = {}, acc3 = {};
        gemm_lds_k1024(xpan + aoff, wp, acc0, acc1, acc2, acc3);
        if (t > 0)
            gemm_lds_k1024(hpan + aoff, wp + Hk, acc0, acc1, acc2, acc3);

        // ---- stage gate tiles in LDS (D layout: VGPR r -> row r, +8 for hi half) ----
        {
            const int rb = hi * 8;
            const int nc = nt * 16 + nl;
#pragma unroll
            for (int r = 0; r < 8; ++r) {
                preact[g][rb + r][nc]      = acc0[r];
                preact[g][rb + 16 + r][nc] = acc1[r];
                preact[g][rb + 32 + r][nc] = acc2[r];
                preact[g][rb + 48 + r][nc] = acc3[r];
            }
        }
        __syncthreads();

        // ---- elementwise gate math + state update (2048 elems / 256 threads) ----
        for (int e = tid; e < 64 * WGCOLS; e += THREADS) {
            const int m = e >> 5, n = e & 31;
            float xi = preact[0][m][n] + bsh[0][n];
            float xf = preact[1][m][n] + bsh[1][n];
            float xg = preact[2][m][n] + bsh[2][n];
            float xo = preact[3][m][n] + bsh[3][n];
            float ig = sigm(xi), fg = sigm(xf), og = sigm(xo);
            float gg = tanhf(xg);
            float c  = cst[m][n] * fg + ig * gg;
            cst[m][n] = c;
            float h = og * tanhf(c);

            hbuf[(size_t)(t & (HMOD - 1)) * BPH + (size_t)m * HP + c0 + n] = (__bf16)h;
            if (FINAL)
                dout[(size_t)t * BHO + (size_t)m * Hk + c0 + n] = h;
            if (t == Tk - 1) {
                size_t o = (size_t)layer * BHO + (size_t)m * Hk + c0 + n;
                dout[(size_t)Tk * BHO + o] = h;                        // output_hs
                dout[(size_t)Tk * BHO + (size_t)Lk * BHO + o] = h;     // output_cs (ref bug: == hs[-1])
            }
        }

        // ---- convert x_{t+1} slice one step ahead (layer 0 only) ----
        if (XSTAGE && (t + 1 < Tk)) {
            const float* xf = xraw + (size_t)(t + 1) * BHO;
            __bf16* xs = xstage_w + (size_t)((t + 1) & 1) * BPH;
            for (int i = tid; i < 2048; i += THREADS) {
                int fl = blockIdx.x * 2048 + i;
                xs[(fl >> 10) * HP + (fl & 1023)] = (__bf16)xf[fl];
            }
        }

        // ---- grid-wide barrier: publish h_t (and x_{t+1}), wait for all WGs ----
        grid_barrier(counter, (++bar) * gridDim.x);
    }
}

extern "C" void kernel_launch(void* const* d_in, const int* in_sizes, int n_in,
                              void* d_out, int out_size, void* d_ws, size_t ws_size,
                              hipStream_t stream) {
    (void)in_sizes; (void)n_in; (void)out_size; (void)ws_size;
    // setup_inputs order: x, h0, c0, i2h_w, i2h_b, h2h_w, h2h_b (h0/c0 are zeros)
    const float* x     = (const float*)d_in[0];
    const float* i2h_w = (const float*)d_in[3];
    const float* i2h_b = (const float*)d_in[4];
    const float* h2h_w = (const float*)d_in[5];
    const float* h2h_b = (const float*)d_in[6];
    float* out = (float*)d_out;

    // Workspace layout (~97.5 MB)
    char* ws = (char*)d_ws;
    size_t off = 0;
    __bf16* hs0    = (__bf16*)(ws + off); off += (size_t)Tk * BPH * sizeof(__bf16);       // layer-0 h seq (padded rows)
    __bf16* hfb    = (__bf16*)(ws + off); off += (size_t)2 * BPH * sizeof(__bf16);        // layer-1 h double buffer
    __bf16* xstage = (__bf16*)(ws + off); off += (size_t)2 * BPH * sizeof(__bf16);        // x bf16 double buffer
    __bf16* wcat   = (__bf16*)(ws + off); off += (size_t)Lk * G4H * K2 * sizeof(__bf16);  // 32 MB packed weights
    float*  bsum   = (float*)(ws + off);  off += (size_t)Lk * G4H * sizeof(float);
    unsigned* counters = (unsigned*)(ws + off);

    const size_t total = (size_t)Lk * G4H * K2;
    const int pblocks = (int)((total + 255) / 256);
    lstm_prep_kernel<<<pblocks, 256, 0, stream>>>(i2h_w, i2h_b, h2h_w, h2h_b, wcat, bsum, counters);

    const size_t dyn_lds = (size_t)2 * Bk * PSTR * sizeof(__bf16);   // 258 KB A panels

    // Layer 0: f32 x via bf16 xstage (converted 1 step ahead), writes hs0 + last-step outputs.
    lstm_layer_kernel<true, false, Tk, 1><<<NWG, THREADS, dyn_lds, stream>>>(
        x, xstage, xstage, wcat, bsum, hs0, out, 0, counters + 0);

    // Layer 1: bf16 hs0 input, writes d_out hs every step + last-step outputs.
    lstm_layer_kernel<false, true, 2, Tk - 1><<<NWG, THREADS, dyn_lds, stream>>>(
        nullptr, hs0, nullptr, wcat + (size_t)G4H * K2, bsum + G4H, hfb, out, 1, counters + 1);
}
